// VllmMixtureOfExpertsOpFP8_66949950210389
// MI455X (gfx1250) — matrix operand order repaired and checked by reading, not compiled, not run on hardware
//
#include <hip/hip_runtime.h>
#include <hip/hip_bf16.h>
#include <cstdint>

// ---------------- problem constants (fixed by the reference) ----------------
#define T_   2048
#define H_   2048
#define I_   1024
#define E_   8
#define TK   2
#define QB   128          // fp8 block-quant tile size
#define CAP  4224         // 4096 routed slots + up to 8*15 padding, 16-aligned
#define MAXTILES (CAP/16) // 264

// ws layout (bytes):
//  [0,128)            : ctrl ints  (0..7 cnt, 8..15 fill, 16..24 aligned offsets, [24]=total)
//  [128, 128+CAP*4)   : tok  (int per slot)
//  [128+CAP*4, +CAP*4): wgt  (float per slot)
//  [128+CAP*8, ...)   : act  (bf16, CAP x I_)  -- 33920 is 16B aligned
#define WS_TOK_OFF 128
#define WS_WGT_OFF (128 + CAP*4)
#define WS_ACT_OFF (128 + CAP*8)

typedef __attribute__((ext_vector_type(16))) __bf16 v16bf;
typedef __attribute__((ext_vector_type(8)))  float  v8f;

union FragB16 { v16bf v; uint32_t u[8]; };

// pack two f32 -> packed bf16 pair (round-half-up) : 2x v_add_u32 + 1x v_perm_b32
static __device__ __forceinline__ uint32_t pkbf(float lo, float hi) {
  uint32_t ul = __float_as_uint(lo) + 0x8000u;
  uint32_t uh = __float_as_uint(hi) + 0x8000u;
  // D = { uh.b3, uh.b2, ul.b3, ul.b2 }  -> (bf16(hi)<<16) | bf16(lo)
  return __builtin_amdgcn_perm(uh, ul, 0x07060302u);
}
static __device__ __forceinline__ uint16_t f2bf1(float f) {
  return (uint16_t)((__float_as_uint(f) + 0x8000u) >> 16);
}

// ---------------- routing ----------------
__global__ void moe_count(const int* __restrict__ ids, int* __restrict__ ctrl, int n) {
  int i = blockIdx.x * blockDim.x + threadIdx.x;
  if (i < n) atomicAdd(&ctrl[ids[i]], 1);
}

__global__ void moe_scan(int* __restrict__ ctrl) {
  if (threadIdx.x == 0 && blockIdx.x == 0) {
    int off = 0;
    for (int e = 0; e < E_; ++e) {
      ctrl[16 + e] = off;
      off += (ctrl[e] + 15) & ~15;   // 16-align each expert's slot range
    }
    ctrl[24] = off;                  // total padded slots
  }
}

__global__ void moe_scatter(const int* __restrict__ ids, const float* __restrict__ tw,
                            int* __restrict__ ctrl, int* __restrict__ tok,
                            float* __restrict__ wgt, int n) {
  int i = blockIdx.x * blockDim.x + threadIdx.x;
  if (i < n) {
    int e = ids[i];
    int slot = ctrl[16 + e] + atomicAdd(&ctrl[8 + e], 1);
    tok[slot] = i / TK;
    wgt[slot] = tw[i];
  }
}

// ---------------- GEMM1: act = silu(x·w13_gate^T)*(x·w13_up^T)*route_w ----------------
#define LS1 132   // bf16 LDS row stride (128 + 4 pad) = 264B rows: b64-store aligned,
                  // 66-word stride spreads banks across the 16 M/N rows

__global__ __launch_bounds__(32)
void moe_gemm1(const float* __restrict__ x, const float* __restrict__ w13,
               const float* __restrict__ s13, const int* __restrict__ ctrl,
               const int* __restrict__ tok, const float* __restrict__ wgt,
               uint16_t* __restrict__ act) {
  __shared__ uint16_t lA[16 * LS1];   // x tile (bf16): 16 tokens x 128 K
  __shared__ uint16_t lB[32 * LS1];   // w13 tile (bf16): 16 gate + 16 up rows x 128 K
  __shared__ int   ltok[16];
  __shared__ float lwgt[16];

  const int lane  = threadIdx.x;
  const int slot0 = blockIdx.y * 16;
  if (slot0 >= ctrl[24]) return;            // uniform exit past routed tiles
  int e = 0;
  #pragma unroll
  for (int i = 1; i < E_; ++i) if (slot0 >= ctrl[16 + i]) e = i;

  const int n1 = blockIdx.x * 16;           // gate column base in [0, I_)
  if (lane < 16) { ltok[lane] = tok[slot0 + lane]; lwgt[lane] = wgt[slot0 + lane]; }
  __syncthreads();

  const float* wbase = w13 + (size_t)e * (2 * I_) * H_;
  const int m  = lane & 15;
  const int hi = lane >> 4;
  v8f cg = {}; v8f cu = {};

  for (int hb = 0; hb < H_ / QB; ++hb) {
    // stage A tile: 16 rows x 32 float4 -> packed bf16 pairs in LDS
    #pragma unroll 4
    for (int it = 0; it < 16; ++it) {
      int lin = it * 32 + lane;
      int row = lin >> 5, c4 = lin & 31;
      const float* src = x + (size_t)ltok[row] * H_ + hb * QB + c4 * 4;
      if (hb + 1 < H_ / QB) __builtin_prefetch(src + QB, 0, 1);
      float4 v = *(const float4*)src;
      uint2 p = { pkbf(v.x, v.y), pkbf(v.z, v.w) };
      *(uint2*)(lA + row * LS1 + c4 * 4) = p;
    }
    // stage B tile: rows 0..15 = gate (n1+r), rows 16..31 = up (I_+n1+r)
    #pragma unroll 4
    for (int it = 0; it < 32; ++it) {
      int lin = it * 32 + lane;
      int row = lin >> 5, c4 = lin & 31;
      int wrow = (row < 16) ? (n1 + row) : (I_ + n1 + row - 16);
      const float* src = wbase + (size_t)wrow * H_ + hb * QB + c4 * 4;
      if (hb + 1 < H_ / QB) __builtin_prefetch(src + QB, 0, 1);
      float4 v = *(const float4*)src;
      uint2 p = { pkbf(v.x, v.y), pkbf(v.z, v.w) };
      *(uint2*)(lB + row * LS1 + c4 * 4) = p;
    }
    __syncthreads();

    v8f bg = {}; v8f bu = {};
    #pragma unroll
    for (int ks = 0; ks < 4; ++ks) {
      FragB16 fa, fg, fu;
      #pragma unroll
      for (int vi = 0; vi < 8; ++vi) {
        // A 16x32 bf16, k order used here (same as B): VGPR vi -> K = 16*laneHi + 2*vi
        int ka = ks * 32 + hi * 16 + 2 * vi;
        fa.u[vi] = *(const uint32_t*)(lA + m * LS1 + ka);
        // B 32x16 bf16, k order used here (same as A): VGPR vi -> K = 16*laneHi + 2*vi
        int kb = ks * 32 + hi * 16 + 2 * vi;
        fg.u[vi] = *(const uint32_t*)(lB + m * LS1 + kb);
        fu.u[vi] = *(const uint32_t*)(lB + (16 + m) * LS1 + kb);
      }
      bg = __builtin_amdgcn_wmma_f32_16x16x32_bf16(false, fa.v, false, fg.v, (short)0, bg, false, false);
      bu = __builtin_amdgcn_wmma_f32_16x16x32_bf16(false, fa.v, false, fu.v, (short)0, bu, false, false);
    }
    // block-dequant: scale this 128-wide K partial before folding into totals
    float sg = s13[((size_t)e * (2 * I_ / QB) + (n1 / QB)) * (H_ / QB) + hb];
    float su = s13[((size_t)e * (2 * I_ / QB) + ((I_ + n1) / QB)) * (H_ / QB) + hb];
    #pragma unroll
    for (int r = 0; r < 8; ++r) { cg[r] += bg[r] * sg; cu[r] += bu[r] * su; }
    __syncthreads();
  }

  // epilogue: act = silu(gate)*up * route_weight (weight folds padding to zero)
  const int n = lane & 15;
  #pragma unroll
  for (int r = 0; r < 8; ++r) {
    int mm = r + 8 * hi;                 // C/D layout: VGPR r -> M = r + 8*laneHi
    float g = cg[r], u = cu[r];
    float a = (g / (1.0f + __expf(-g))) * u * lwgt[mm];
    act[(size_t)(slot0 + mm) * I_ + n1 + n] = f2bf1(a);
  }
}

// ---------------- GEMM2: out[t] += act · w2^T (block-dequant), atomic combine ----------------
#define LSA2 136  // bf16 LDS row stride (128 + 8 pad) -> 272B rows, b128-store aligned

__global__ __launch_bounds__(32)
void moe_gemm2(const uint16_t* __restrict__ act, const float* __restrict__ w2,
               const float* __restrict__ s2, const int* __restrict__ ctrl,
               const int* __restrict__ tok, float* __restrict__ out) {
  __shared__ uint16_t lA[16 * LSA2];  // act tile: 16 slots x 128 K (bf16, copied)
  __shared__ uint16_t lB[16 * LS1];   // w2 tile: 16 h-rows x 128 K (bf16, converted)
  __shared__ int      ltok[16];

  const int lane  = threadIdx.x;
  const int slot0 = blockIdx.y * 16;
  if (slot0 >= ctrl[24]) return;
  int e = 0;
  #pragma unroll
  for (int i = 1; i < E_; ++i) if (slot0 >= ctrl[16 + i]) e = i;

  const int h1 = blockIdx.x * 16;
  if (lane < 16) ltok[lane] = tok[slot0 + lane];
  __syncthreads();

  const float* wbase = w2 + (size_t)e * H_ * I_;
  const int m  = lane & 15;
  const int hi = lane >> 4;
  v8f c = {};

  for (int ib = 0; ib < I_ / QB; ++ib) {
    // stage A (bf16 already): 16 rows x 8 uint4, straight b128 copies
    #pragma unroll
    for (int it = 0; it < 4; ++it) {
      int lin = it * 32 + lane;
      int row = lin >> 3, c16 = lin & 7;
      const uint16_t* src = act + (size_t)(slot0 + row) * I_ + ib * QB + c16 * 8;
      *(uint4*)(lA + row * LSA2 + c16 * 8) = *(const uint4*)src;
    }
    // stage B (fp32 -> packed bf16): 16 rows x 32 float4
    #pragma unroll 4
    for (int it = 0; it < 16; ++it) {
      int lin = it * 32 + lane;
      int row = lin >> 5, c4 = lin & 31;
      const float* src = wbase + (size_t)(h1 + row) * I_ + ib * QB + c4 * 4;
      if (ib + 1 < I_ / QB) __builtin_prefetch(src + QB, 0, 1);
      float4 v = *(const float4*)src;
      uint2 p = { pkbf(v.x, v.y), pkbf(v.z, v.w) };
      *(uint2*)(lB + row * LS1 + c4 * 4) = p;
    }
    __syncthreads();

    v8f cb = {};
    #pragma unroll
    for (int ks = 0; ks < 4; ++ks) {
      FragB16 fa, fb;
      #pragma unroll
      for (int vi = 0; vi < 8; ++vi) {
        int ka = ks * 32 + hi * 16 + 2 * vi;
        fa.u[vi] = *(const uint32_t*)(lA + m * LSA2 + ka);
        int kb = ks * 32 + hi * 16 + 2 * vi;
        fb.u[vi] = *(const uint32_t*)(lB + m * LS1 + kb);
      }
      cb = __builtin_amdgcn_wmma_f32_16x16x32_bf16(false, fa.v, false, fb.v, (short)0, cb, false, false);
    }
    float s = s2[((size_t)e * (H_ / QB) + (h1 / QB)) * (I_ / QB) + ib];
    #pragma unroll
    for (int r = 0; r < 8; ++r) c[r] += cb[r] * s;
    __syncthreads();
  }

  const int n = lane & 15;
  #pragma unroll
  for (int r = 0; r < 8; ++r) {
    int mm = r + 8 * hi;
    atomicAdd(out + (size_t)ltok[mm] * H_ + h1 + n, c[r]);
  }
}

// ---------------- host launch ----------------
extern "C" void kernel_launch(void* const* d_in, const int* in_sizes, int n_in,
                              void* d_out, int out_size, void* d_ws, size_t ws_size,
                              hipStream_t stream) {
  (void)in_sizes; (void)n_in; (void)out_size; (void)ws_size;
  const float* x   = (const float*)d_in[0];
  const int*   ids = (const int*)d_in[1];
  const float* tw  = (const float*)d_in[2];
  const float* w13 = (const float*)d_in[3];
  const float* s13 = (const float*)d_in[4];
  const float* w2  = (const float*)d_in[5];
  const float* s2  = (const float*)d_in[6];
  float* out = (float*)d_out;

  int*      ctrl = (int*)d_ws;
  int*      tok  = (int*)((char*)d_ws + WS_TOK_OFF);
  float*    wgt  = (float*)((char*)d_ws + WS_WGT_OFF);
  uint16_t* actb = (uint16_t*)((char*)d_ws + WS_ACT_OFF);

  // zero routing state + output (capture-safe async memsets)
  hipMemsetAsync(d_ws, 0, WS_ACT_OFF, stream);
  hipMemsetAsync(d_out, 0, (size_t)T_ * H_ * sizeof(float), stream);

  const int n = T_ * TK;
  moe_count  <<<(n + 255) / 256, 256, 0, stream>>>(ids, ctrl, n);
  moe_scan   <<<1, 32, 0, stream>>>(ctrl);
  moe_scatter<<<(n + 255) / 256, 256, 0, stream>>>(ids, tw, ctrl, tok, wgt, n);

  moe_gemm1<<<dim3(I_ / 16, MAXTILES), 32, 0, stream>>>(x, w13, s13, ctrl, tok, wgt, actb);
  moe_gemm2<<<dim3(H_ / 16, MAXTILES), 32, 0, stream>>>(actb, w2, s2, ctrl, tok, out);
}
